// LMUFFTCell_1082331758664
// MI455X (gfx1250) — matrix operand order repaired and checked by reading, not compiled, not run on hardware
//
#include <hip/hip_runtime.h>

#define S_LEN  4096
#define IN_SZ  256
#define MEM_SZ 256
#define HID    512
#define NBATCH 16

typedef __attribute__((ext_vector_type(16))) _Float16     v16h;
typedef __attribute__((ext_vector_type(8)))  float        v8f;
typedef __attribute__((ext_vector_type(8)))  unsigned int v8u;
typedef __attribute__((ext_vector_type(4)))  unsigned int u32x4;
typedef __attribute__((ext_vector_type(8)))  int          i32x8;
typedef __attribute__((ext_vector_type(4)))  int          i32x4;

static __device__ __forceinline__ unsigned int pk2(float a, float b) {
  auto p = __builtin_amdgcn_cvt_pkrtz(a, b);          // v_cvt_pk_rtz_f16_f32
  return __builtin_bit_cast(unsigned int, p);
}
static __device__ __forceinline__ unsigned short f2h(float a) {
  _Float16 h = (_Float16)a;
  return __builtin_bit_cast(unsigned short, h);
}
// A-fragment (16x32 f16) K pair-base for element pair j (0..7) per lane half.
// Contiguous order: lanes 0-15 hold K 0..15 ; lanes 16-31 hold K 16..31.
static __device__ __forceinline__ int a_kbase(int j, int half) {
  return 2 * j + 16 * half;
}

// ---------------- Stage 1: u = relu(x @ Wu^T + b), stored f16 ----------------
__global__ void __launch_bounds__(256) lmu_u_kernel(
    const float* __restrict__ x, const float* __restrict__ wu,
    const float* __restrict__ wub, unsigned short* __restrict__ u16) {
  const int wave = threadIdx.x >> 5, lane = threadIdx.x & 31;
  const int r = blockIdx.x * 8 + wave;                // one wave per row
  const float* xr = x + (size_t)r * IN_SZ;
  float s = 0.f;
#pragma unroll
  for (int i = 0; i < IN_SZ / 32; ++i) {
    const int c = lane + i * 32;
    s += xr[c] * wu[c];
  }
#pragma unroll
  for (int off = 16; off; off >>= 1) s += __shfl_xor(s, off, 32);
  if (lane == 0) u16[r] = f2h(fmaxf(s + wub[0], 0.f));
}

// ---------------- Stage 2: convert H and Wh to f16 ----------------
__global__ void lmu_cvt_kernel(const float* __restrict__ H,
                               const float* __restrict__ Wh,
                               unsigned short* __restrict__ H16,
                               unsigned short* __restrict__ Wh16) {
  const int nH = MEM_SZ * S_LEN;           // 1048576
  const int nW = HID * (MEM_SZ + IN_SZ);   // 262144
  const int i = blockIdx.x * blockDim.x + threadIdx.x;
  const int stride = gridDim.x * blockDim.x;
  for (int k = i; k < nH; k += stride) H16[k] = f2h(H[k]);
  for (int k = i; k < nW; k += stride) Wh16[k] = f2h(Wh[k]);
}

// ---------------- Stage 3: causal conv as Toeplitz WMMA matmul ----------------
// Block = (batch b, 32-wide time tile); 8 waves each own 32 mem rows.
// C[mem,t] += A[mem,d] * B[d,t],  A = H rows (global, L2-hot),
// B[d,t] = u_pad[t-d].  u is stored REVERSED in LDS (two parity-shifted
// copies) so every K-pair of the B fragment is one aligned ds_load_b32.
__global__ void __launch_bounds__(256) lmu_conv_kernel(
    const unsigned short* __restrict__ u16,
    const unsigned short* __restrict__ H16,
    unsigned short* __restrict__ m16) {
  __shared__ unsigned short uRa[4160];  // uRa[i] = u_pad[L - i]
  __shared__ unsigned short uRb[4160];  // uRb[i] = u_pad[L - i - 1]
  const int b = blockIdx.x >> 7;
  const int tb = blockIdx.x & 127;
  const int t0 = tb << 5;
  const int tid = threadIdx.x;

  {
    const unsigned short* ub = u16 + (size_t)b * S_LEN;
    const int L = t0 + 31;
    const int tot = t0 + 64;
    for (int i = tid; i < tot; i += 256) {
      const int ia = L - i, ib = ia - 1;
      uRa[i] = (ia >= 0) ? ub[ia] : (unsigned short)0;
      uRb[i] = (ib >= 0) ? ub[ib] : (unsigned short)0;
    }
  }
  __syncthreads();

  const int wave = tid >> 5, lane = tid & 31;
  const int half = lane >> 4, lm = lane & 15;
  const int mem_base = wave << 5;          // 8 waves cover 256 mem rows
  // lane-parity base pointer: elements for lane are uR[base + e], base parity
  // = parity of (31 - lm); odd lm -> even base (uRa), even lm -> uRb shifted.
  const unsigned short* up = (lm & 1) ? uRa : (uRb - 1);
  v8f zero = {};
  v8f acc[2][2];
#pragma unroll
  for (int i = 0; i < 2; ++i)
#pragma unroll
    for (int j = 0; j < 2; ++j) acc[i][j] = zero;

  const int nchunk = tb + 1;               // causal: only d <= t contributes
  for (int kc = 0; kc < nchunk; ++kc) {
    const int d0 = kc << 5;
    v16h afrag[2];
#pragma unroll
    for (int mt = 0; mt < 2; ++mt) {
      const unsigned short* hrow =
          H16 + (size_t)(mem_base + mt * 16 + lm) * S_LEN + d0;
      v8u au;
#pragma unroll
      for (int j = 0; j < 8; ++j)
        au[j] = *(const unsigned int*)(hrow + a_kbase(j, half));
      afrag[mt] = __builtin_bit_cast(v16h, au);
      if (kc + 1 < nchunk) __builtin_prefetch(hrow + 32, 0, 1);
    }
    v16h bfrag[2];
#pragma unroll
    for (int tt = 0; tt < 2; ++tt) {
      const int base = 31 - tt * 16 - lm + d0 + 16 * half;
      v8u bu;
#pragma unroll
      for (int j = 0; j < 8; ++j)
        bu[j] = *(const unsigned int*)(up + base + 2 * j);
      bfrag[tt] = __builtin_bit_cast(v16h, bu);
    }
#pragma unroll
    for (int mt = 0; mt < 2; ++mt)
#pragma unroll
      for (int tt = 0; tt < 2; ++tt)
        acc[mt][tt] = __builtin_amdgcn_wmma_f32_16x16x32_f16(
            false, afrag[mt], false, bfrag[tt], (short)0, acc[mt][tt],
            false, false);
  }

  // store m16[b][t][mem], packing f32 pairs -> f16 dwords
#pragma unroll
  for (int mt = 0; mt < 2; ++mt)
#pragma unroll
    for (int tt = 0; tt < 2; ++tt) {
      const int t_col = t0 + tt * 16 + lm;
      unsigned int* mrow =
          (unsigned int*)(m16 + ((size_t)(b * S_LEN + t_col)) * MEM_SZ +
                          mem_base + mt * 16 + 8 * half);
#pragma unroll
      for (int v = 0; v < 8; v += 2)
        mrow[v >> 1] = pk2(acc[mt][tt][v], acc[mt][tt][v + 1]);
    }
}

// ---------------- Stage 4: h = relu([m, x] @ Wh^T + b) ----------------
// Block = 128 rows x 64 cols; Wh tile (64x512 f16 = 64KB) staged into LDS
// by the Tensor Data Mover (D# per ISA ch.8) when available.
__global__ void __launch_bounds__(256) lmu_h_kernel(
    const float* __restrict__ x, const unsigned short* __restrict__ m16,
    const unsigned short* __restrict__ Wh16, const float* __restrict__ whb,
    float* __restrict__ h, float* __restrict__ hlast) {
  __shared__ unsigned short whS[64 * 512];
  const int rb = blockIdx.x >> 3, cbk = blockIdx.x & 7;
  const int r0 = rb << 7;    // 128 rows (never straddles a batch)
  const int n0 = cbk << 6;   // 64 cols
  const int tid = threadIdx.x;

#if defined(__HIP_DEVICE_COMPILE__) && \
    __has_builtin(__builtin_amdgcn_tensor_load_to_lds) && \
    __has_builtin(__builtin_amdgcn_s_wait_tensorcnt)
  if (tid < 32) {  // one TDM issue per block (TDM ignores EXEC)
    const unsigned long long ga =
        (unsigned long long)(const void*)(Wh16 + (size_t)n0 * 512);
    const unsigned int lds = (unsigned int)(unsigned long long)(void*)whS;
    // D# group0: count=1, lds_addr, global_addr[56:0], type=2
    u32x4 g0 = {1u, lds, (unsigned int)ga,
                (unsigned int)((ga >> 32) & 0x1FFFFFFu) | (2u << 30)};
    // D# group1: data_size=2B; tensor_dim0=512, tensor_dim1=64;
    // tile_dim0=512, tile_dim1=64; tensor_dim0_stride=512
    i32x8 g1 = {(int)(1u << 16), (int)(512u << 16), (int)(64u << 16),
                (int)(512u << 16), 64, 512, 0, 0};
    i32x4 gz = {0, 0, 0, 0};
#if __clang_major__ >= 23
    i32x8 gz8 = {0, 0, 0, 0, 0, 0, 0, 0};
    __builtin_amdgcn_tensor_load_to_lds(g0, g1, gz, gz, gz8, 0);
#else
    __builtin_amdgcn_tensor_load_to_lds(g0, g1, gz, gz, 0);
#endif
    __builtin_amdgcn_s_wait_tensorcnt(0);
  }
#else
  {
    unsigned int* d = (unsigned int*)whS;
    const unsigned int* s = (const unsigned int*)(Wh16 + (size_t)n0 * 512);
    for (int i = tid; i < 64 * 512 / 2; i += 256) d[i] = s[i];
  }
#endif
  __syncthreads();

  const int wave = tid >> 5, lane = tid & 31;
  const int half = lane >> 4, lm = lane & 15;
  const int wrow = wave >> 1, wcol = wave & 1;
  const int rbase = r0 + wrow * 32;
  const int nbase_l = wcol * 32;
  v8f zero = {};
  v8f acc[2][2];
#pragma unroll
  for (int i = 0; i < 2; ++i)
#pragma unroll
    for (int j = 0; j < 2; ++j) acc[i][j] = zero;

  // K = 0..255 : memory features m (f16 in workspace)
  for (int kc = 0; kc < 8; ++kc) {
    const int k0 = kc << 5;
    v16h afrag[2];
#pragma unroll
    for (int mt = 0; mt < 2; ++mt) {
      const unsigned short* arow =
          m16 + (size_t)(rbase + mt * 16 + lm) * MEM_SZ + k0;
      v8u au;
#pragma unroll
      for (int j = 0; j < 8; ++j)
        au[j] = *(const unsigned int*)(arow + a_kbase(j, half));
      afrag[mt] = __builtin_bit_cast(v16h, au);
    }
    v16h bfrag[2];
#pragma unroll
    for (int nt = 0; nt < 2; ++nt) {
      const int base = (nbase_l + nt * 16 + lm) * 512 + k0 + 16 * half;
      v8u bu;
#pragma unroll
      for (int j = 0; j < 8; ++j)
        bu[j] = *(const unsigned int*)(whS + base + 2 * j);
      bfrag[nt] = __builtin_bit_cast(v16h, bu);
    }
#pragma unroll
    for (int mt = 0; mt < 2; ++mt)
#pragma unroll
      for (int nt = 0; nt < 2; ++nt)
        acc[mt][nt] = __builtin_amdgcn_wmma_f32_16x16x32_f16(
            false, afrag[mt], false, bfrag[nt], (short)0, acc[mt][nt],
            false, false);
  }
  // K = 256..511 : raw inputs x (f32, converted on the fly)
  for (int kc = 0; kc < 8; ++kc) {
    const int k0 = kc << 5;
    v16h afrag[2];
#pragma unroll
    for (int mt = 0; mt < 2; ++mt) {
      const float* xrow = x + (size_t)(rbase + mt * 16 + lm) * IN_SZ + k0;
      v8u au;
#pragma unroll
      for (int j = 0; j < 8; ++j) {
        const int K = a_kbase(j, half);
        au[j] = pk2(xrow[K], xrow[K + 1]);
      }
      afrag[mt] = __builtin_bit_cast(v16h, au);
    }
    v16h bfrag[2];
#pragma unroll
    for (int nt = 0; nt < 2; ++nt) {
      const int base = (nbase_l + nt * 16 + lm) * 512 + 256 + k0 + 16 * half;
      v8u bu;
#pragma unroll
      for (int j = 0; j < 8; ++j)
        bu[j] = *(const unsigned int*)(whS + base + 2 * j);
      bfrag[nt] = __builtin_bit_cast(v16h, bu);
    }
#pragma unroll
    for (int mt = 0; mt < 2; ++mt)
#pragma unroll
      for (int nt = 0; nt < 2; ++nt)
        acc[mt][nt] = __builtin_amdgcn_wmma_f32_16x16x32_f16(
            false, afrag[mt], false, bfrag[nt], (short)0, acc[mt][nt],
            false, false);
  }

  // epilogue: bias + relu, write h and h[:, -1, :]
#pragma unroll
  for (int mt = 0; mt < 2; ++mt)
#pragma unroll
    for (int nt = 0; nt < 2; ++nt) {
      const int n_g = n0 + nbase_l + nt * 16 + lm;
      const float bias = whb[n_g];
#pragma unroll
      for (int v = 0; v < 8; ++v) {
        const int r_g = rbase + mt * 16 + 8 * half + v;
        const float val = fmaxf(acc[mt][nt][v] + bias, 0.f);
        h[(size_t)r_g * HID + n_g] = val;
        if ((r_g & (S_LEN - 1)) == S_LEN - 1)
          hlast[(size_t)(r_g >> 12) * HID + n_g] = val;
      }
    }
}

extern "C" void kernel_launch(void* const* d_in, const int* in_sizes, int n_in,
                              void* d_out, int out_size, void* d_ws,
                              size_t ws_size, hipStream_t stream) {
  const float* x   = (const float*)d_in[0];
  const float* wuw = (const float*)d_in[1];
  const float* wub = (const float*)d_in[2];
  const float* whw = (const float*)d_in[3];
  const float* whb = (const float*)d_in[4];
  const float* H   = (const float*)d_in[5];

  char* ws = (char*)d_ws;
  unsigned short* u16  = (unsigned short*)(ws);                         // 128 KB
  unsigned short* H16  = (unsigned short*)(ws + (1u << 17));            // 2 MB
  unsigned short* Wh16 = (unsigned short*)(ws + (1u << 17) + (1u << 21));        // 512 KB
  unsigned short* m16  = (unsigned short*)(ws + (1u << 17) + (1u << 21) + (1u << 19)); // 32 MB

  float* h = (float*)d_out;
  float* hlast = h + (size_t)NBATCH * S_LEN * HID;

  lmu_u_kernel<<<(NBATCH * S_LEN) / 8, 256, 0, stream>>>(x, wuw, wub, u16);
  lmu_cvt_kernel<<<2048, 256, 0, stream>>>(H, whw, H16, Wh16);
  lmu_conv_kernel<<<NBATCH * (S_LEN / 32), 256, 0, stream>>>(u16, H16, m16);
  lmu_h_kernel<<<(NBATCH * S_LEN / 128) * (HID / 64), 256, 0, stream>>>(
      x, m16, Wh16, whb, h, hlast);
}
